// MHA_9706626089289
// MI455X (gfx1250) — matrix operand order repaired and checked by reading, not compiled, not run on hardware
//
#include <hip/hip_runtime.h>

// MHA for MI455X (gfx1250, wave32, WMMA).
// N=4096 tokens, E=512, H=8, DH=64. All matmuls on v_wmma_f32_16x16x32_f16.
// ~43 GFLOP total vs ~26 MB f16 working set (L2-resident at 192 MB) ->
// compute-bound; raise arithmetic intensity per fragment (32x64 wave tiles:
// every B fragment feeds 2 WMMAs) and double-buffer k-step fragments so
// in-order VMEM completion overlaps WMMA issue.

#define N_TOK 4096
#define E_DIM 512
#define H_NUM 8
#define DH    64

typedef __attribute__((ext_vector_type(16))) _Float16 v16h;
typedef __attribute__((ext_vector_type(8)))  _Float16 v8h;
typedef __attribute__((ext_vector_type(8)))  float    v8f;

static __device__ __forceinline__ v8f wmma16(v16h a, v16h b, v8f c) {
  // D = A(16x32 f16) * B(32x16 f16) + C(16x16 f32)
  return __builtin_amdgcn_wmma_f32_16x16x32_f16(false, a, false, b, (short)0, c,
                                                false, false);
}

// ---- Fragment loaders (CDNA5 ISA 7.12.2 layouts) -------------------------
// A fragment 16x32 f16: lane l&15 holds row M; VGPR0..3 = K[bk..bk+8),
// VGPR4..7 = K[16+bk..16+bk+8), bk = 8 for lanes 16..31.
static __device__ __forceinline__ v16h load_fragA(const _Float16* base, int ld) {
  const int l   = threadIdx.x & 31;
  const int row = l & 15;
  const int bk  = (l >> 4) << 3;                 // 0 or 8
  const _Float16* p = base + row * ld;
  v8h lo = *(const v8h*)(p + bk);
  v8h hi = *(const v8h*)(p + bk + 16);
  v16h f;
#pragma unroll
  for (int i = 0; i < 8; ++i) { f[i] = lo[i]; f[i + 8] = hi[i]; }
  return f;
}

// B fragment 32x16 f16: lane holds column N=l&15; VGPR0..3 = K[bk..bk+8),
// VGPR4..7 = K[16+bk..16+bk+8), bk = 8 for lanes 16..31. Source is B^T row-major (B columns K-contiguous).
static __device__ __forceinline__ v16h load_fragB(const _Float16* base, int ld) {
  const int l   = threadIdx.x & 31;
  const int col = l & 15;
  const int bk  = (l >> 4) << 3;                 // 0 or 8
  const _Float16* p = base + col * ld + bk;
  v8h lo = *(const v8h*)(p);
  v8h hi = *(const v8h*)(p + 16);
  v16h f;
#pragma unroll
  for (int i = 0; i < 8; ++i) { f[i] = lo[i]; f[i + 8] = hi[i]; }
  return f;
}

// ---- Double-buffered 32x64 GEMM core: acc += A(32x512) * B(512x64) -------
// Arow = &A[n0*512], Bcol = &B[o0*512] (row-major, ld = 512).
// acc[0..3] = rows n0..n0+16, acc[4..7] = rows n0+16..n0+32, col tiles t*16.
// 12 b128 loads feed 8 WMMAs per k-step; next-step fragments are loaded
// before the current WMMAs so the (in-order) VMEM pipe overlaps WMMA issue.
static __device__ __forceinline__ void gemm_32x64(const _Float16* __restrict__ Arow,
                                                  const _Float16* __restrict__ Bcol,
                                                  v8f acc[8]) {
  v16h a0 = load_fragA(Arow, E_DIM);
  v16h a1 = load_fragA(Arow + 16 * E_DIM, E_DIM);
  v16h b0 = load_fragB(Bcol + 0 * E_DIM,  E_DIM);
  v16h b1 = load_fragB(Bcol + 16 * E_DIM, E_DIM);
  v16h b2 = load_fragB(Bcol + 32 * E_DIM, E_DIM);
  v16h b3 = load_fragB(Bcol + 48 * E_DIM, E_DIM);
#pragma unroll 2
  for (int k0 = 32; k0 < E_DIM; k0 += 32) {
    v16h a0n = load_fragA(Arow + k0, E_DIM);
    v16h a1n = load_fragA(Arow + 16 * E_DIM + k0, E_DIM);
    v16h bn0 = load_fragB(Bcol + 0 * E_DIM + k0,  E_DIM);
    v16h bn1 = load_fragB(Bcol + 16 * E_DIM + k0, E_DIM);
    v16h bn2 = load_fragB(Bcol + 32 * E_DIM + k0, E_DIM);
    v16h bn3 = load_fragB(Bcol + 48 * E_DIM + k0, E_DIM);
    acc[0] = wmma16(a0, b0, acc[0]);
    acc[4] = wmma16(a1, b0, acc[4]);
    acc[1] = wmma16(a0, b1, acc[1]);
    acc[5] = wmma16(a1, b1, acc[5]);
    acc[2] = wmma16(a0, b2, acc[2]);
    acc[6] = wmma16(a1, b2, acc[6]);
    acc[3] = wmma16(a0, b3, acc[3]);
    acc[7] = wmma16(a1, b3, acc[7]);
    a0 = a0n; a1 = a1n; b0 = bn0; b1 = bn1; b2 = bn2; b3 = bn3;
  }
  acc[0] = wmma16(a0, b0, acc[0]);
  acc[4] = wmma16(a1, b0, acc[4]);
  acc[1] = wmma16(a0, b1, acc[1]);
  acc[5] = wmma16(a1, b1, acc[5]);
  acc[2] = wmma16(a0, b2, acc[2]);
  acc[6] = wmma16(a1, b2, acc[6]);
  acc[3] = wmma16(a0, b3, acc[3]);
  acc[7] = wmma16(a1, b3, acc[7]);
}

// ---- f32 -> f16 convert (grid-stride) ------------------------------------
__global__ void k_f32_to_f16(const float* __restrict__ src,
                             _Float16* __restrict__ dst, int n) {
  for (int i = blockIdx.x * blockDim.x + threadIdx.x; i < n;
       i += gridDim.x * blockDim.x)
    dst[i] = (_Float16)src[i];
}

// ---- QKV projection: Y[n,o] = sum_i X[n,i]*W[o,i] + b[o] -----------------
// blockIdx.z: 0=K, 1=Q, 2=V. Scatter-stores into the reference's reshapes.
// 4 waves/block, each wave a 32x64 tile -> block covers 128 rows x 64 cols.
__global__ void k_qkv_gemm(const _Float16* __restrict__ X,
                           const _Float16* __restrict__ WK,
                           const _Float16* __restrict__ WQ,
                           const _Float16* __restrict__ WV,
                           const float* __restrict__ bK,
                           const float* __restrict__ bQ,
                           const float* __restrict__ bV,
                           _Float16* __restrict__ Kh,   // [H][4096][64]
                           _Float16* __restrict__ Qh,   // [H][4096][64] (pre /DH)
                           _Float16* __restrict__ Vt) { // [H][64][4096]
  const int wave = threadIdx.x >> 5;
  const int lane = threadIdx.x & 31;
  const int n0   = blockIdx.x * 128 + wave * 32;
  const int o0   = blockIdx.y * 64;
  const int mode = blockIdx.z;

  const _Float16* W  = (mode == 0) ? WK : (mode == 1) ? WQ : WV;
  const float*  bias = (mode == 0) ? bK : (mode == 1) ? bQ : bV;

  v8f acc[8] = {};
  gemm_32x64(X + (size_t)n0 * E_DIM, W + (size_t)o0 * E_DIM, acc);

  const int cw = lane & 15;
  const int rb = (lane >> 4) << 3;               // 0 or 8
#pragma unroll
  for (int u = 0; u < 2; ++u) {
#pragma unroll
    for (int t = 0; t < 4; ++t) {
      const int o  = o0 + t * 16 + cw;
      const float bo = bias[o];
#pragma unroll
      for (int r = 0; r < 8; ++r) {
        const int n = n0 + u * 16 + rb + r;
        const float y = acc[u * 4 + t][r] + bo;
        if (mode == 0) {
          // K: transpose(.,(2,1,0)).reshape(H,N,DH)
          const int h = n >> 9, m = ((n & 511) << 3) + (o >> 6), d = o & 63;
          Kh[(((size_t)h << 12) + m) * DH + d] = (_Float16)y;
        } else {
          // Q/V: raw (E,N) memory reshaped to (H,N,DH)
          const int h = o >> 6, n2 = ((o & 63) << 6) + (n >> 6), d = n & 63;
          if (mode == 1)
            Qh[(((size_t)h << 12) + n2) * DH + d] = (_Float16)(y * (1.0f / DH));
          else
            Vt[((size_t)h * DH + d) * N_TOK + n2] = (_Float16)y;
        }
      }
    }
  }
}

// ---- Flash attention: per wave = 16 query rows, stream 32 keys / iter ----
__global__ void k_attention(const _Float16* __restrict__ Qh,
                            const _Float16* __restrict__ Kh,
                            const _Float16* __restrict__ Vt,
                            _Float16* __restrict__ Z16) { // [N][E] = [n'][h*64+d]
  __shared__ __align__(16) _Float16 Plds[4][16][40];      // 80B pitch (16B mult)
  const int wave = threadIdx.x >> 5;
  const int lane = threadIdx.x & 31;
  const int h    = blockIdx.y;
  const int n0   = blockIdx.x * 64 + wave * 16;
  _Float16 (*P)[40] = Plds[wave];

  const _Float16* Qb = Qh + (((size_t)h << 12) + n0) * DH;
  const _Float16* Kb = Kh + ((size_t)h << 12) * DH;
  const _Float16* Vb = Vt + (size_t)h * DH * N_TOK;

  const v16h qa0 = load_fragA(Qb, DH);       // d = 0..31
  const v16h qa1 = load_fragA(Qb + 32, DH);  // d = 32..63

  v8f o0 = {}, o1 = {}, o2 = {}, o3 = {};
  float rm[8], rl[8];
#pragma unroll
  for (int r = 0; r < 8; ++r) { rm[r] = -1e30f; rl[r] = 0.0f; }

  const int cw = lane & 15;
  const int rb = (lane >> 4) << 3;

  for (int m0 = 0; m0 < N_TOK; m0 += 32) {
    if (m0 + 32 < N_TOK) {
      __builtin_prefetch(Kb + (size_t)(m0 + 32) * DH, 0, 1);
      __builtin_prefetch(Vb + m0 + 32, 0, 1);
    }
    // ---- Issue ALL K and V fragment loads up front. K loads (older)
    // complete first and gate the S-WMMAs with a partial wait; V loads stay
    // in flight beneath the softmax VALU work and drain at the PV-WMMAs.
    v16h kb00 = load_fragB(Kb + (size_t)m0 * DH, DH);
    v16h kb01 = load_fragB(Kb + (size_t)m0 * DH + 32, DH);
    v16h kb10 = load_fragB(Kb + (size_t)(m0 + 16) * DH, DH);
    v16h kb11 = load_fragB(Kb + (size_t)(m0 + 16) * DH + 32, DH);
    v16h vb0  = load_fragB(Vb + (size_t)0 * N_TOK + m0,  N_TOK);
    v16h vb1  = load_fragB(Vb + (size_t)16 * N_TOK + m0, N_TOK);
    v16h vb2  = load_fragB(Vb + (size_t)32 * N_TOK + m0, N_TOK);
    v16h vb3  = load_fragB(Vb + (size_t)48 * N_TOK + m0, N_TOK);

    // ---- S = (Q/DH) * K^T, two 16x16 tiles (32 keys) ----
    v8f s0 = {}, s1 = {};
    s0 = wmma16(qa0, kb00, s0); s0 = wmma16(qa1, kb01, s0);
    s1 = wmma16(qa0, kb10, s1); s1 = wmma16(qa1, kb11, s1);

    // ---- online softmax (row stats live per (VGPR r, 16-lane half)) ----
    float scale[8];
#pragma unroll
    for (int r = 0; r < 8; ++r) {
      float mx = fmaxf(s0[r], s1[r]);
      mx = fmaxf(mx, __shfl_xor(mx, 1, 32));
      mx = fmaxf(mx, __shfl_xor(mx, 2, 32));
      mx = fmaxf(mx, __shfl_xor(mx, 4, 32));
      mx = fmaxf(mx, __shfl_xor(mx, 8, 32));
      const float mnew = fmaxf(rm[r], mx);
      const float p0 = __expf(s0[r] - mnew);
      const float p1 = __expf(s1[r] - mnew);
      const float sc = __expf(rm[r] - mnew);
      rm[r] = mnew;
      float ps = p0 + p1;
      ps += __shfl_xor(ps, 1, 32);
      ps += __shfl_xor(ps, 2, 32);
      ps += __shfl_xor(ps, 4, 32);
      ps += __shfl_xor(ps, 8, 32);
      rl[r] = rl[r] * sc + ps;
      scale[r] = sc;
      s0[r] = p0; s1[r] = p1;
    }
#pragma unroll
    for (int r = 0; r < 8; ++r) {
      o0[r] *= scale[r]; o1[r] *= scale[r];
      o2[r] *= scale[r]; o3[r] *= scale[r];
    }

    // ---- P (C-layout f32) -> LDS -> A-fragment f16 (16x32) ----
#pragma unroll
    for (int r = 0; r < 8; ++r) {
      P[rb + r][cw]      = (_Float16)s0[r];
      P[rb + r][16 + cw] = (_Float16)s1[r];
    }
    asm volatile("s_wait_dscnt 0x0" ::: "memory");  // per-wave DS in-order fence
    const v16h pa = load_fragA(&P[0][0], 40);

    // ---- O += P * V (V fragments already resident) ----
    o0 = wmma16(pa, vb0, o0);
    o1 = wmma16(pa, vb1, o1);
    o2 = wmma16(pa, vb2, o2);
    o3 = wmma16(pa, vb3, o3);
  }

  float inv[8];
#pragma unroll
  for (int r = 0; r < 8; ++r) inv[r] = 1.0f / rl[r];
#pragma unroll
  for (int r = 0; r < 8; ++r) {
    _Float16* zr = Z16 + (size_t)(n0 + rb + r) * E_DIM + h * DH;
    zr[cw]      = (_Float16)(o0[r] * inv[r]);
    zr[16 + cw] = (_Float16)(o1[r] * inv[r]);
    zr[32 + cw] = (_Float16)(o2[r] * inv[r]);
    zr[48 + cw] = (_Float16)(o3[r] * inv[r]);
  }
}

// ---- Output projection: out[n,e] = sum_i Z[n,i]*WZ[e,i] + bZ[e] (f32) ----
__global__ void k_out_gemm(const _Float16* __restrict__ Z,
                           const _Float16* __restrict__ WZ,
                           const float* __restrict__ bZ,
                           float* __restrict__ out) {
  const int wave = threadIdx.x >> 5;
  const int lane = threadIdx.x & 31;
  const int n0   = blockIdx.x * 128 + wave * 32;
  const int o0   = blockIdx.y * 64;

  v8f acc[8] = {};
  gemm_32x64(Z + (size_t)n0 * E_DIM, WZ + (size_t)o0 * E_DIM, acc);

  const int cw = lane & 15;
  const int rb = (lane >> 4) << 3;
#pragma unroll
  for (int u = 0; u < 2; ++u) {
#pragma unroll
    for (int t = 0; t < 4; ++t) {
      const int o = o0 + t * 16 + cw;
      const float bo = bZ[o];
#pragma unroll
      for (int r = 0; r < 8; ++r)
        out[(size_t)(n0 + u * 16 + rb + r) * E_DIM + o] = acc[u * 4 + t][r] + bo;
    }
  }
}

extern "C" void kernel_launch(void* const* d_in, const int* in_sizes, int n_in,
                              void* d_out, int out_size, void* d_ws,
                              size_t ws_size, hipStream_t stream) {
  (void)in_sizes; (void)n_in; (void)out_size; (void)ws_size;
  const float* inp  = (const float*)d_in[0];
  const float* WKw  = (const float*)d_in[1];
  const float* WKb  = (const float*)d_in[2];
  const float* WQw  = (const float*)d_in[3];
  const float* WQb  = (const float*)d_in[4];
  const float* WVw  = (const float*)d_in[5];
  const float* WVb  = (const float*)d_in[6];
  const float* WZw  = (const float*)d_in[7];
  const float* WZb  = (const float*)d_in[8];

  const size_t NE = (size_t)N_TOK * E_DIM;   // 2,097,152
  const size_t EE = (size_t)E_DIM * E_DIM;   //   262,144
  char* ws = (char*)d_ws;
  _Float16* X16  = (_Float16*)ws;            ws += NE * 2;
  _Float16* WK16 = (_Float16*)ws;            ws += EE * 2;
  _Float16* WQ16 = (_Float16*)ws;            ws += EE * 2;
  _Float16* WV16 = (_Float16*)ws;            ws += EE * 2;
  _Float16* WZ16 = (_Float16*)ws;            ws += EE * 2;
  _Float16* Qh   = (_Float16*)ws;            ws += NE * 2;
  _Float16* Kh   = (_Float16*)ws;            ws += NE * 2;
  _Float16* Vt   = (_Float16*)ws;            ws += NE * 2;
  _Float16* Z16  = (_Float16*)ws;            ws += NE * 2;

  k_f32_to_f16<<<2048, 256, 0, stream>>>(inp, X16, (int)NE);
  k_f32_to_f16<<<512,  256, 0, stream>>>(WKw, WK16, (int)EE);
  k_f32_to_f16<<<512,  256, 0, stream>>>(WQw, WQ16, (int)EE);
  k_f32_to_f16<<<512,  256, 0, stream>>>(WVw, WV16, (int)EE);
  k_f32_to_f16<<<512,  256, 0, stream>>>(WZw, WZ16, (int)EE);

  k_qkv_gemm<<<dim3(N_TOK / 128, E_DIM / 64, 3), 128, 0, stream>>>(
      X16, WK16, WQ16, WV16, WKb, WQb, WVb, Kh, Qh, Vt);

  k_attention<<<dim3(N_TOK / 64, H_NUM), 128, 0, stream>>>(Qh, Kh, Vt, Z16);

  k_out_gemm<<<dim3(N_TOK / 128, E_DIM / 64), 128, 0, stream>>>(
      Z16, WZ16, WZb, (float*)d_out);
}
